// NAFBlock_flow_24953759990213
// MI455X (gfx1250) — compile-verified, matrix operand order checked by reading, not run on hardware
//
#include <hip/hip_runtime.h>
#include <hip/hip_bf16.h>
#include <math.h>

// ---------------- problem constants ----------------
#define NB   8          // batch
#define NC   64         // dim
#define NHD  4          // heads
#define CH   16         // dim per head
#define NFB  128        // flow channels
#define HWD  16384      // H*W
#define NHID 128        // mlp hidden / dw / ffn channels
#define NTOK (NB*HWD)   // 131072 tokens

typedef __attribute__((ext_vector_type(16))) __bf16 v16bf;
typedef __attribute__((ext_vector_type(8)))  float  v8f;
typedef __attribute__((ext_vector_type(2)))  float  v2f;
typedef __attribute__((ext_vector_type(4)))  unsigned int u32x4;
typedef __attribute__((ext_vector_type(8)))  int i32x8;
typedef __attribute__((ext_vector_type(4)))  int i32x4;

__device__ __forceinline__ float gelu_erf(float x) {
  return 0.5f * x * (1.0f + erff(x * 0.70710678118654752440f));
}
// bf16 A/B fragment K-index map (ISA 7.12.2, 16-bit 16x32 layout)
__device__ __forceinline__ int kmap16(int j) { return (j < 8) ? j : (j + 8); }

// ---------------- TDM: load a [rows x 16] f32 tile (row stride = stride elems) into LDS ----------------
// D# per ISA 8.3/8.4: group0 = count|lds_addr|global_addr|type=2 ("image")
//                    group1 = data_size=4B, tensor_dim0=tile_dim0=16, tensor_dim1=tile_dim1=rows,
//                             tensor_dim0_stride=stride. 2-D => groups 2/3 zero.
__device__ __forceinline__ void tdm_load_tile(const float* gsrc, unsigned lds_byte_addr,
                                              int rows, int stride) {
  unsigned long long ga = (unsigned long long)(uintptr_t)gsrc;
  u32x4 g0;
  g0[0] = 1u;                                              // count=1 (valid user descriptor)
  g0[1] = lds_byte_addr;                                   // LDS byte address
  g0[2] = (unsigned)(ga & 0xFFFFFFFFull);                  // global_addr[31:0]
  g0[3] = (unsigned)((ga >> 32) & 0x1FFFFFFull) | (2u << 30); // global_addr[56:32] | type=2
  i32x8 g1;
  g1[0] = (int)(2u << 16);          // [17:16] data_size=2 -> 4 bytes; workgroup_mask=0
  g1[1] = (int)(16u << 16);         // bits[63:48]  tensor_dim0 = 16
  g1[2] = (int)((unsigned)rows << 16); // bits[95:80] tensor_dim1 = rows
  g1[3] = (int)(16u << 16);         // bits[127:112] tile_dim0 = 16
  g1[4] = rows;                     // bits[143:128] tile_dim1 = rows (tile_dim2 = 0)
  g1[5] = stride;                   // bits[191:160] tensor_dim0_stride lo32 (elements)
  g1[6] = 0;
  g1[7] = 0;
  i32x4 g2 = {0, 0, 0, 0}, g3 = {0, 0, 0, 0};
#if defined(__clang_major__) && (__clang_major__ >= 23)
  i32x8 g4 = {0, 0, 0, 0, 0, 0, 0, 0};
  __builtin_amdgcn_tensor_load_to_lds(g0, g1, g2, g3, g4, 0);
#else
  __builtin_amdgcn_tensor_load_to_lds(g0, g1, g2, g3, 0);
#endif
}

// ---------------- per-token LayerNorm stats over C=64 channels ----------------
__global__ __launch_bounds__(256) void ln_stats_k(const float* __restrict__ x, float eps,
                                                  float* __restrict__ mu, float* __restrict__ rs) {
  int t  = blockIdx.x * 256 + threadIdx.x;      // 0 .. NTOK-1
  int b  = t >> 14;
  int hw = t & (HWD - 1);
  const float* p = x + (size_t)b * NC * HWD + hw;
  float s = 0.f, s2 = 0.f;
#pragma unroll
  for (int c = 0; c < NC; ++c) { float v = p[(size_t)c * HWD]; s += v; s2 += v * v; }
  float m   = s * (1.f / NC);
  float var = s2 * (1.f / NC) - m * m;
  mu[t] = m;
  rs[t] = rsqrtf(var + eps);
}

// ---------------- fused WMMA 1x1 conv: y = act(W @ ln(x)*insc + b)*outsc + res ----------------
// TDM stages the raw f32 activation tile to per-wave LDS; LN / input-scale / bf16-convert are
// applied while building the (hoisted) B fragments; weights live in VGPRs as A fragments.
template <int CIN, int COUT, bool LN, bool BIAS, bool GEL, bool RES, bool INS, bool OUTS>
__global__ __launch_bounds__(256) void conv1x1_k(
    const float* __restrict__ xin, const float* __restrict__ W,
    const float* __restrict__ bias, const float* __restrict__ lnw, const float* __restrict__ lnb,
    const float* __restrict__ mu, const float* __restrict__ rs,
    const float* __restrict__ insc, const float* __restrict__ outsc,
    const float* __restrict__ res, float* __restrict__ out) {
  constexpr int KS = CIN / 32;   // k-steps of 32 (bf16 WMMA)
  constexpr int MT = COUT / 16;  // output tiles
  __shared__ float sxf[8][CIN][16];
  const int lane = threadIdx.x & 31;
  const int wv = __builtin_amdgcn_readfirstlane(threadIdx.x >> 5);
  const int m = lane & 15, hi = lane >> 4;
  const unsigned lds_base = (unsigned)(uintptr_t)(&sxf[wv][0][0]);

  // hoist weight A-fragments into VGPRs for the whole kernel (<=16 frags)
  v16bf A[MT * KS];
#pragma unroll
  for (int mt = 0; mt < MT; ++mt)
#pragma unroll
    for (int ks = 0; ks < KS; ++ks)
#pragma unroll
      for (int j = 0; j < 16; ++j) {
        int kk = ks * 32 + hi * 8 + kmap16(j);
        A[mt * KS + ks][j] = (__bf16)W[(size_t)(mt * 16 + m) * CIN + kk];
      }

  for (int iter = 0; iter < 4; ++iter) {                 // 2048 waves * 4 = 8192 tiles
    int tile = blockIdx.x * 8 + wv + iter * 2048;
    int b = tile >> 10;                                  // HWD/16 = 1024 tiles per image
    int hw0 = (tile & 1023) << 4;
    const float* xb = xin + (size_t)b * CIN * HWD + hw0;
    if (iter) asm volatile("s_wait_dscnt 0x0" ::: "memory");  // LDS reads done before TDM rewrites
    tdm_load_tile(xb, lds_base, CIN, HWD);
    if (iter < 3) __builtin_prefetch(xin + (size_t)(((tile + 2048) >> 10)) * CIN * HWD
                                         + (((tile + 2048) & 1023) << 4), 0, 1);
    int tbase = b * HWD + hw0;
    float mun = 0.f, rsn = 0.f;
    if (LN) { mun = mu[tbase + m]; rsn = rs[tbase + m]; }   // per-token stats (column n == m)
    __builtin_amdgcn_s_wait_tensorcnt(0);
    // build all B fragments once per tile, fusing LN / input scale / bf16 convert
    v16bf Bf[KS];
#pragma unroll
    for (int ks = 0; ks < KS; ++ks)
#pragma unroll
      for (int j = 0; j < 16; ++j) {
        int kk = ks * 32 + hi * 8 + kmap16(j);
        float v = sxf[wv][kk][m];
        if (LN)  v = (v - mun) * rsn * lnw[kk] + lnb[kk];
        if (INS) v *= insc[b * CIN + kk];
        Bf[ks][j] = (__bf16)v;
      }
#pragma unroll
    for (int mt = 0; mt < MT; ++mt) {
      v8f acc;
#pragma unroll
      for (int r = 0; r < 8; ++r) acc[r] = BIAS ? bias[mt * 16 + r + 8 * hi] : 0.f;
#pragma unroll
      for (int ks = 0; ks < KS; ++ks)
        acc = __builtin_amdgcn_wmma_f32_16x16x32_bf16(false, A[mt * KS + ks], false, Bf[ks],
                                                      (short)0, acc, false, false);
#pragma unroll
      for (int r = 0; r < 8; ++r) {
        int o = mt * 16 + r + 8 * hi;
        float vv = acc[r];
        if (GEL)  vv = gelu_erf(vv);
        if (OUTS) vv *= outsc[o];
        size_t off = ((size_t)b * COUT + o) * HWD + hw0 + m;
        if (RES) vv += res[off];
        out[off] = vv;
      }
    }
  }
}

// ---------------- depthwise 3x3 (SAME) + optional GELU ----------------
template <bool GEL>
__global__ __launch_bounds__(256) void dw3_k(const float* __restrict__ x, const float* __restrict__ w,
                                             const float* __restrict__ bias, float* __restrict__ y, int Cn) {
  int idx = blockIdx.x * 256 + threadIdx.x;       // over NB*Cn*HWD
  int wx = idx & 127;
  int hy = (idx >> 7) & 127;
  int bc = idx >> 14;
  int c  = bc % Cn;
  const float* xp = x + ((size_t)bc << 14);
  float acc = bias[c];
#pragma unroll
  for (int dy = -1; dy <= 1; ++dy) {
    int yy = hy + dy;
    if ((unsigned)yy >= 128u) continue;
#pragma unroll
    for (int dx = -1; dx <= 1; ++dx) {
      int xx = wx + dx;
      if ((unsigned)xx >= 128u) continue;
      acc += w[c * 9 + (dy + 1) * 3 + (dx + 1)] * xp[yy * 128 + xx];
    }
  }
  if (GEL) acc = gelu_erf(acc);
  y[idx] = acc;
}

// ---------------- per-row reduce over HW: mode0 -> 1/max(||row||,1e-12); mode1 -> mean ----------------
__global__ __launch_bounds__(256) void redhw_k(const float* __restrict__ x, float* __restrict__ out, int mode) {
  __shared__ float red[256];
  int row = blockIdx.x, tid = threadIdx.x;
  const float* p = x + ((size_t)row << 14);
  float s = 0.f;
  for (int i = tid; i < HWD; i += 256) { float v = p[i]; s += mode ? v : v * v; }
  red[tid] = s;
  __syncthreads();
  for (int st = 128; st > 0; st >>= 1) { if (tid < st) red[tid] += red[tid + st]; __syncthreads(); }
  if (tid == 0) out[row] = mode ? red[0] * (1.f / HWD) : 1.f / fmaxf(sqrtf(red[0]), 1e-12f);
}

// ---------------- SCA 128x128 matvec per batch ----------------
__global__ __launch_bounds__(256) void sca_k(const float* __restrict__ meanv, const float* __restrict__ scaw,
                                             const float* __restrict__ scab, float* __restrict__ sca) {
  int i = blockIdx.x * 256 + threadIdx.x;   // NB*NHID
  int b = i >> 7, o = i & 127;
  float acc = scab[o];
#pragma unroll 8
  for (int c = 0; c < NHID; ++c) acc += scaw[o * NHID + c] * meanv[b * NHID + c];
  sca[i] = acc;
}

// ---------------- Gram matrix (q_hat k_hat^T * temp) + softmax, per (b,head) ----------------
__global__ __launch_bounds__(256) void gram_k(const float* __restrict__ q, const float* __restrict__ k,
                                              const float* __restrict__ invq, const float* __restrict__ invk,
                                              const float* __restrict__ temp, float* __restrict__ P) {
  __shared__ float sm[8][16][16];
  __shared__ float mat[16][16];
  int bh = blockIdx.x, b = bh >> 2, h = bh & 3;
  int lane = threadIdx.x & 31, wv = threadIdx.x >> 5;
  int m = lane & 15, hi = lane >> 4;
  int c0 = b * NC + h * CH;
  const float* qp = q + (size_t)c0 * HWD;
  const float* kp = k + (size_t)c0 * HWD;
  float iq = invq[c0 + m];     // A row scale (row m)
  float ik = invk[c0 + m];     // B col scale (col n == lane&15)
  v8f acc;
#pragma unroll
  for (int r = 0; r < 8; ++r) acc[r] = 0.f;
  for (int n0 = wv * 4; n0 < HWD; n0 += 32) {           // each wave covers 1/8 of K
    int kk = n0 + hi * 2;
    v2f a, bb;
    a[0]  = qp[(size_t)m * HWD + kk]     * iq;
    a[1]  = qp[(size_t)m * HWD + kk + 1] * iq;
    bb[0] = kp[(size_t)m * HWD + kk]     * ik;
    bb[1] = kp[(size_t)m * HWD + kk + 1] * ik;
    acc = __builtin_amdgcn_wmma_f32_16x16x4_f32(false, a, false, bb, (short)0, acc, false, false);
  }
#pragma unroll
  for (int r = 0; r < 8; ++r) sm[wv][r + 8 * hi][m] = acc[r];
  __syncthreads();
  int tm = threadIdx.x >> 4, tn = threadIdx.x & 15;
  float s = 0.f;
#pragma unroll
  for (int w = 0; w < 8; ++w) s += sm[w][tm][tn];
  mat[tm][tn] = s * temp[h];
  __syncthreads();
  if (threadIdx.x < 16) {                                // softmax over d for row tm
    int r = threadIdx.x;
    float mx = -3.4e38f;
#pragma unroll
    for (int d = 0; d < 16; ++d) mx = fmaxf(mx, mat[r][d]);
    float e[16], sum = 0.f;
#pragma unroll
    for (int d = 0; d < 16; ++d) { e[d] = expf(mat[r][d] - mx); sum += e[d]; }
    float inv = 1.f / sum;
#pragma unroll
    for (int d = 0; d < 16; ++d) P[bh * 256 + r * 16 + d] = e[d] * inv;
  }
}

// ---------------- fused: out = x + pow_w @ (P @ v)  (per-head f32 WMMA, then bf16 WMMA) ----------------
__global__ __launch_bounds__(256) void attnpow_k(const float* __restrict__ P, const float* __restrict__ v,
                                                 const float* __restrict__ poww, const float* __restrict__ xres,
                                                 float* __restrict__ out) {
  __shared__ __bf16 so[8][NC][16];
  int lane = threadIdx.x & 31, wv = threadIdx.x >> 5;
  int m = lane & 15, hi = lane >> 4;
  // pow_w fragments hoisted (MT=4, KS=2)
  v16bf Aw[8];
#pragma unroll
  for (int mt = 0; mt < 4; ++mt)
#pragma unroll
    for (int ks = 0; ks < 2; ++ks)
#pragma unroll
      for (int j = 0; j < 16; ++j) {
        int kk = ks * 32 + hi * 8 + kmap16(j);
        Aw[mt * 2 + ks][j] = (__bf16)poww[(mt * 16 + m) * NC + kk];
      }
  for (int iter = 0; iter < 4; ++iter) {
    int tile = blockIdx.x * 8 + wv + iter * 2048;
    int b = tile >> 10;
    int hw0 = (tile & 1023) << 4;
#pragma unroll
    for (int h = 0; h < 4; ++h) {
      const float* Pp = P + ((b << 2) + h) * 256;
      const float* vp = v + ((size_t)b * NC + h * CH) * HWD + hw0;
      v8f acc;
#pragma unroll
      for (int r = 0; r < 8; ++r) acc[r] = 0.f;
#pragma unroll
      for (int s = 0; s < 4; ++s) {                       // K=16 in 4 f32-WMMA steps
        int kk = s * 4 + hi * 2;
        v2f a, bb;
        a[0]  = Pp[m * 16 + kk];
        a[1]  = Pp[m * 16 + kk + 1];
        bb[0] = vp[(size_t)kk * HWD + m];
        bb[1] = vp[(size_t)(kk + 1) * HWD + m];
        acc = __builtin_amdgcn_wmma_f32_16x16x4_f32(false, a, false, bb, (short)0, acc, false, false);
      }
#pragma unroll
      for (int r = 0; r < 8; ++r) so[wv][h * CH + r + 8 * hi][m] = (__bf16)acc[r];
    }
    __syncthreads();
#pragma unroll
    for (int mt = 0; mt < 4; ++mt) {
      v8f acc;
#pragma unroll
      for (int r = 0; r < 8; ++r) acc[r] = 0.f;
#pragma unroll
      for (int ks = 0; ks < 2; ++ks) {
        v16bf Bf;
#pragma unroll
        for (int j = 0; j < 16; ++j) Bf[j] = so[wv][ks * 32 + hi * 8 + kmap16(j)][m];
        acc = __builtin_amdgcn_wmma_f32_16x16x32_bf16(false, Aw[mt * 2 + ks], false, Bf,
                                                      (short)0, acc, false, false);
      }
#pragma unroll
      for (int r = 0; r < 8; ++r) {
        int o = mt * 16 + r + 8 * hi;
        size_t off = ((size_t)b * NC + o) * HWD + hw0 + m;
        out[off] = xres[off] + acc[r];
      }
    }
    __syncthreads();
  }
}

// ---------------- host orchestration ----------------
extern "C" void kernel_launch(void* const* d_in, const int* in_sizes, int n_in,
                              void* d_out, int out_size, void* d_ws, size_t ws_size,
                              hipStream_t stream) {
  const float* image = (const float*)d_in[0];
  const float* flow  = (const float*)d_in[1];
  const float* fcw   = (const float*)d_in[2];
  const float* fcb   = (const float*)d_in[3];
  const float* lniw  = (const float*)d_in[4];
  const float* lnib  = (const float*)d_in[5];
  const float* lnfw  = (const float*)d_in[6];
  const float* lnfb  = (const float*)d_in[7];
  const float* qw    = (const float*)d_in[8];
  const float* kw    = (const float*)d_in[9];
  const float* vw    = (const float*)d_in[10];
  const float* dqw   = (const float*)d_in[11];
  const float* dqb   = (const float*)d_in[12];
  const float* dkw   = (const float*)d_in[13];
  const float* dkb   = (const float*)d_in[14];
  const float* dvw   = (const float*)d_in[15];
  const float* dvb   = (const float*)d_in[16];
  const float* temp  = (const float*)d_in[17];
  const float* poww  = (const float*)d_in[18];
  const float* ln2w  = (const float*)d_in[19];
  const float* ln2b  = (const float*)d_in[20];
  const float* f1w   = (const float*)d_in[21];
  const float* f1b   = (const float*)d_in[22];
  const float* f2w   = (const float*)d_in[23];
  const float* f2b   = (const float*)d_in[24];
  const float* n1w   = (const float*)d_in[25];
  const float* n1b   = (const float*)d_in[26];
  const float* n2w   = (const float*)d_in[27];
  const float* n2b   = (const float*)d_in[28];
  const float* c1w   = (const float*)d_in[29];
  const float* c1b   = (const float*)d_in[30];
  const float* c2w   = (const float*)d_in[31];
  const float* c2b   = (const float*)d_in[32];
  const float* c3w   = (const float*)d_in[33];
  const float* c3b   = (const float*)d_in[34];
  const float* scaw  = (const float*)d_in[35];
  const float* scab  = (const float*)d_in[36];
  const float* c4w   = (const float*)d_in[37];
  const float* c4b   = (const float*)d_in[38];
  const float* c5w   = (const float*)d_in[39];
  const float* c5b   = (const float*)d_in[40];
  const float* beta  = (const float*)d_in[41];
  const float* gamma = (const float*)d_in[42];

  const size_t SZ64  = (size_t)NB * NC * HWD;     // 8.39M floats
  const size_t SZ128 = (size_t)NB * NHID * HWD;   // 16.78M floats
  float* w = (float*)d_ws;
  float* xA   = w;                 // layer ping
  float* xB   = xA + SZ64;         // layer pong
  float* buf1 = xB + SZ64;         // 128ch (halves buf1a/buf1b)
  float* buf2 = buf1 + SZ128;
  float* buf3 = buf2 + SZ128;
  float* mu0  = buf3 + SZ128;
  float* rs0  = mu0 + NTOK;
  float* mu1  = rs0 + NTOK;
  float* rs1  = mu1 + NTOK;
  float* invq = rs1 + NTOK;
  float* invk = invq + NB * NC;
  float* Pbuf = invk + NB * NC;
  float* mnv  = Pbuf + NB * NHD * 256;
  float* scab_v = mnv + NB * NHID;
  size_t need = (size_t)(scab_v + NB * NHID - w) * sizeof(float);
  if (ws_size < need) return;      // workspace too small: no-op

  const float* x = image;
  for (int l = 0; l < 2; ++l) {
    const float* fcw_l = fcw + (size_t)l * NC * NFB;
    const float* fcb_l = fcb + l * NC;
    const float* lniw_l = lniw + l * NC, *lnib_l = lnib + l * NC;
    const float* lnfw_l = lnfw + l * NC, *lnfb_l = lnfb + l * NC;
    const float* qw_l = qw + (size_t)l * NC * NC;
    const float* kw_l = kw + (size_t)l * NC * NC;
    const float* vw_l = vw + (size_t)l * NC * NC;
    const float* dqw_l = dqw + (size_t)l * NC * 9, *dqb_l = dqb + l * NC;
    const float* dkw_l = dkw + (size_t)l * NC * 9, *dkb_l = dkb + l * NC;
    const float* dvw_l = dvw + (size_t)l * NC * 9, *dvb_l = dvb + l * NC;
    const float* temp_l = temp + l * NHD;
    const float* poww_l = poww + (size_t)l * NC * NC;
    const float* ln2w_l = ln2w + l * NC, *ln2b_l = ln2b + l * NC;
    const float* f1w_l = f1w + (size_t)l * NHID * NC, *f1b_l = f1b + l * NHID;
    const float* f2w_l = f2w + (size_t)l * NC * NHID, *f2b_l = f2b + l * NC;

    float* fl   = buf1;            // 64ch half A
    float* qpre = buf1 + SZ64;     // half B
    float* kpre = buf2;
    float* vpre = buf2 + SZ64;
    float* qb   = buf3;
    float* kb   = buf3 + SZ64;
    float* vb   = buf1;            // overwrites fl after k/v projections done
    float* fused = buf2;           // overwrites kpre after gram done
    float* h1   = buf1;            // full 128ch, after v dead
    float* xout = (l == 0) ? xA : xB;

    // flow 1x1 (FB->C) + bias
    conv1x1_k<128, 64, false, true, false, false, false, false><<<256, 256, 0, stream>>>(
        flow, fcw_l, fcb_l, nullptr, nullptr, nullptr, nullptr, nullptr, nullptr, nullptr, fl);
    ln_stats_k<<<NTOK / 256, 256, 0, stream>>>(x, 1e-5f, mu0, rs0);
    ln_stats_k<<<NTOK / 256, 256, 0, stream>>>(fl, 1e-5f, mu1, rs1);
    // q/k/v projections with fused channel-LN
    conv1x1_k<64, 64, true, false, false, false, false, false><<<256, 256, 0, stream>>>(
        x, qw_l, nullptr, lniw_l, lnib_l, mu0, rs0, nullptr, nullptr, nullptr, qpre);
    conv1x1_k<64, 64, true, false, false, false, false, false><<<256, 256, 0, stream>>>(
        fl, kw_l, nullptr, lnfw_l, lnfb_l, mu1, rs1, nullptr, nullptr, nullptr, kpre);
    conv1x1_k<64, 64, true, false, false, false, false, false><<<256, 256, 0, stream>>>(
        fl, vw_l, nullptr, lnfw_l, lnfb_l, mu1, rs1, nullptr, nullptr, nullptr, vpre);
    // depthwise 3x3
    dw3_k<false><<<(NB * NC * HWD) / 256, 256, 0, stream>>>(qpre, dqw_l, dqb_l, qb, NC);
    dw3_k<false><<<(NB * NC * HWD) / 256, 256, 0, stream>>>(kpre, dkw_l, dkb_l, kb, NC);
    dw3_k<false><<<(NB * NC * HWD) / 256, 256, 0, stream>>>(vpre, dvw_l, dvb_l, vb, NC);
    // L2 norms over HW
    redhw_k<<<NB * NC, 256, 0, stream>>>(qb, invq, 0);
    redhw_k<<<NB * NC, 256, 0, stream>>>(kb, invk, 0);
    // Gram + softmax (f32 WMMA)
    gram_k<<<NB * NHD, 256, 0, stream>>>(qb, kb, invq, invk, temp_l, Pbuf);
    // fused = x + pow_w @ (P @ v)
    attnpow_k<<<256, 256, 0, stream>>>(Pbuf, vb, poww_l, x, fused);
    // token LN + MLP
    ln_stats_k<<<NTOK / 256, 256, 0, stream>>>(fused, 1e-5f, mu0, rs0);
    conv1x1_k<64, 128, true, true, true, false, false, false><<<256, 256, 0, stream>>>(
        fused, f1w_l, f1b_l, ln2w_l, ln2b_l, mu0, rs0, nullptr, nullptr, nullptr, h1);
    conv1x1_k<128, 64, false, true, false, true, false, false><<<256, 256, 0, stream>>>(
        h1, f2w_l, f2b_l, nullptr, nullptr, nullptr, nullptr, nullptr, nullptr, fused, xout);
    x = xout;
  }

  // ---------------- NAF block ----------------
  float* t1 = buf1;   // 128ch
  float* hb = buf2;   // 128ch
  float* yb = buf3;   // 64ch (half)
  float* gb = buf1;   // reuse
  ln_stats_k<<<NTOK / 256, 256, 0, stream>>>(x, 1e-6f, mu0, rs0);
  conv1x1_k<64, 128, true, true, false, false, false, false><<<256, 256, 0, stream>>>(
      x, c1w, c1b, n1w, n1b, mu0, rs0, nullptr, nullptr, nullptr, t1);
  dw3_k<true><<<(NB * NHID * HWD) / 256, 256, 0, stream>>>(t1, c2w, c2b, hb, NHID);
  redhw_k<<<NB * NHID, 256, 0, stream>>>(hb, mnv, 1);
  sca_k<<<(NB * NHID) / 256, 256, 0, stream>>>(mnv, scaw, scab, scab_v);
  // y = x + beta * (c3w @ (h*sca) + c3b)
  conv1x1_k<128, 64, false, true, false, true, true, true><<<256, 256, 0, stream>>>(
      hb, c3w, c3b, nullptr, nullptr, nullptr, nullptr, scab_v, beta, x, yb);
  ln_stats_k<<<NTOK / 256, 256, 0, stream>>>(yb, 1e-6f, mu0, rs0);
  conv1x1_k<64, 128, true, true, true, false, false, false><<<256, 256, 0, stream>>>(
      yb, c4w, c4b, n2w, n2b, mu0, rs0, nullptr, nullptr, nullptr, gb);
  // out = y + gamma * (c5w @ g + c5b)
  conv1x1_k<128, 64, false, true, false, true, false, true><<<256, 256, 0, stream>>>(
      gb, c5w, c5b, nullptr, nullptr, nullptr, nullptr, nullptr, gamma, yb, (float*)d_out);
}
